// Attention_76656576299165
// MI455X (gfx1250) — compile-verified, matrix operand order checked by reading, not run on hardware
//
#include <hip/hip_runtime.h>

// ---------------------------------------------------------------------------
// Nystromformer attention forward for gfx1250 (MI455X), wave32 + WMMA bf16,
// async global->LDS staging (ASYNCcnt) with double buffering; 64x64 wave
// tiles (16 WMMAs per 16 LDS fragment loads).
// ---------------------------------------------------------------------------

typedef __bf16 bf16_t;
typedef __attribute__((ext_vector_type(16))) __bf16 v16bf;
typedef __attribute__((ext_vector_type(8)))  float  v8f;

#define SEQ  4096
#define NH   16
#define HD   64
#define NLM  256
#define NBAT 4
#define BH   (NBAT*NH)   // 64 batched heads

union FragB16 { uint4 q[2]; v16bf v; };

__device__ inline v8f vzero8() {
  v8f z;
  for (int i = 0; i < 8; ++i) z[i] = 0.0f;
  return z;
}

__device__ inline v8f wmma_bf(v16bf a, v16bf b, v8f c) {
  return __builtin_amdgcn_wmma_f32_16x16x32_bf16(false, a, false, b,
                                                 (short)0, c, false, false);
}

// Async 16B global -> LDS copy (per-lane). lds = low 32 bits of flat shared
// address == LDS byte offset (ISA 10.2).
__device__ inline void async_b128(const void* lds_ptr, const void* gptr) {
  unsigned lds = (unsigned)(size_t)lds_ptr;
  unsigned long long g = (unsigned long long)gptr;
  asm volatile("global_load_async_to_lds_b128 %0, %1, off"
               :: "v"(lds), "v"(g) : "memory");
}
__device__ inline void wait_async0() {
  asm volatile("s_wait_asynccnt 0x0" ::: "memory");
}

// ---------------------------------------------------------------------------
// Generic bf16 NT GEMM: C[M x N] = A[M x K] * Bt[N x K]^T, batched via z.
// Block = 256 threads (8 waves), 128x256 tile, K-step 32.
// Wave grid 2(M) x 4(N); each wave owns a 64x64 sub-tile = 4x4 WMMA tiles.
// Non-ATRANS path: double-buffered async global->LDS tile staging.
// Epilogues:
//   0: bf16 [B,H,S,D] store of (acc+bias)*mask*scale          (Q / K proj)
//   1: bf16 [B,H,D,S] transposed store of (acc+bias)          (V proj)
//   2: f32 row-major store of acc+bias                        (final proj)
//   3: bf16 raw k3 scores with -1e9*(1-mask[col]) additive    (k3 scores)
//   4: f32 store of acc * 1/rowsum(stats)                     (k3 @ V)
// ATRANS: A elements become exp(x - rowmax(stats)) while staging (reg path).
// ---------------------------------------------------------------------------
template <int EPI, bool ATRANS>
__global__ __launch_bounds__(256) void gemm_bf16_nt(
    const bf16_t* __restrict__ A, const bf16_t* __restrict__ Bt,
    int N, int K, int lda, int ldb,
    long sA, long sB, long sOut,
    const float* __restrict__ bias, const float* __restrict__ mask,
    const float* __restrict__ stats,
    float* __restrict__ outF, bf16_t* __restrict__ outB,
    float scaleMul)
{
  __shared__ bf16_t As[2][128][40];   // pitch 80B: all 16B chunks aligned
  __shared__ bf16_t Bs[2][256][40];

  const int tid = threadIdx.x;
  const int l = tid & 31, w = tid >> 5;
  const int wm = w >> 2, wn = w & 3;         // 2 x 4 wave grid
  const int g = l >> 4, rl = l & 15;
  const int bm = blockIdx.y, bn = blockIdx.x, bz = blockIdx.z;

  A  += (long)bz * sA;
  Bt += (long)bz * sB;

  v8f acc[4][4];
#pragma unroll
  for (int mi = 0; mi < 4; ++mi)
#pragma unroll
    for (int ni = 0; ni < 4; ++ni) acc[mi][ni] = vzero8();

  const int r2 = tid >> 1;               // A staging row (0..127)
  const int seg = (tid & 1) * 16;        // A staging 16-elem half
  const int gnB = bn * 256 + tid;        // B staging row (one per thread)

  // stage one K-tile (A: 2 x b128 per thread, B: 4 x b128 per thread)
  auto stageAB = [&](int buf, int k0) {
    const bf16_t* ga = A + (long)(bm * 128 + r2) * lda + k0 + seg;
    __builtin_prefetch(ga + 64, 0, 1);       // +2 K-tiles into GL2
    async_b128(&As[buf][r2][seg], ga);
    async_b128(&As[buf][r2][seg + 8], ga + 8);
    if (gnB < N) {
      const bf16_t* gb = Bt + (long)gnB * ldb + k0;
#pragma unroll
      for (int c = 0; c < 4; ++c)
        async_b128(&Bs[buf][tid][c * 8], gb + c * 8);
    } else {
      uint4 z; z.x = z.y = z.z = z.w = 0u;
#pragma unroll
      for (int c = 0; c < 2; ++c)
        *(uint4*)&Bs[buf][tid][c * 16] = z, *(uint4*)&Bs[buf][tid][c * 16 + 8] = z;
    }
  };

  auto compute = [&](int buf) {
    v16bf afr[4], bfr[4];
#pragma unroll
    for (int mi = 0; mi < 4; ++mi) {
      FragB16 f;
      f.q[0] = *(const uint4*)&As[buf][wm * 64 + mi * 16 + rl][g * 8];
      f.q[1] = *(const uint4*)&As[buf][wm * 64 + mi * 16 + rl][16 + g * 8];
      afr[mi] = f.v;
    }
#pragma unroll
    for (int ni = 0; ni < 4; ++ni) {
      FragB16 f;
      f.q[0] = *(const uint4*)&Bs[buf][wn * 64 + ni * 16 + rl][g * 8];
      f.q[1] = *(const uint4*)&Bs[buf][wn * 64 + ni * 16 + rl][16 + g * 8];
      bfr[ni] = f.v;
    }
#pragma unroll
    for (int mi = 0; mi < 4; ++mi)
#pragma unroll
      for (int ni = 0; ni < 4; ++ni)
        acc[mi][ni] = wmma_bf(afr[mi], bfr[ni], acc[mi][ni]);
  };

  if (!ATRANS) {
    // ---------------- async, double-buffered pipeline ----------------
    const int nk = K / 32;
    stageAB(0, 0);
    wait_async0();
    __syncthreads();
    for (int ki = 0; ki < nk; ++ki) {
      const int cur = ki & 1;
      if (ki + 1 < nk) stageAB(cur ^ 1, (ki + 1) * 32);
      compute(cur);
      wait_async0();           // next tile landed in LDS
      __syncthreads();         // everyone done reading cur, async visible
    }
  } else {
    // ---------------- ATRANS: A transformed in registers, B async ----------
    for (int k0 = 0; k0 < K; k0 += 32) {
      __syncthreads();
      {
        const bf16_t* ga = A + (long)(bm * 128 + r2) * lda + k0 + seg;
        float mx = stats[((long)bz * 256 + bm * 128 + r2) * 2];
        bf16_t tmp[16];
#pragma unroll
        for (int i = 0; i < 16; ++i)
          tmp[i] = (bf16_t)__expf((float)ga[i] - mx);
        *(uint4*)&As[0][r2][seg]     = *(const uint4*)&tmp[0];
        *(uint4*)&As[0][r2][seg + 8] = *(const uint4*)&tmp[8];
      }
      if (gnB < N) {
        const bf16_t* gb = Bt + (long)gnB * ldb + k0;
#pragma unroll
        for (int c = 0; c < 4; ++c)
          async_b128(&Bs[0][tid][c * 8], gb + c * 8);
      } else {
        uint4 z; z.x = z.y = z.z = z.w = 0u;
#pragma unroll
        for (int c = 0; c < 2; ++c)
          *(uint4*)&Bs[0][tid][c * 16] = z, *(uint4*)&Bs[0][tid][c * 16 + 8] = z;
      }
      wait_async0();
      __syncthreads();
      compute(0);
    }
  }

  // ---- epilogue ----
#pragma unroll
  for (int mi = 0; mi < 4; ++mi) {
#pragma unroll
    for (int ni = 0; ni < 4; ++ni) {
#pragma unroll
      for (int e = 0; e < 8; ++e) {
        int rr = bm * 128 + wm * 64 + mi * 16 + g * 8 + e;
        int cc = bn * 256 + wn * 64 + ni * 16 + rl;
        if (cc >= N) continue;
        float v = acc[mi][ni][e];
        if (EPI == 0) {          // Q/K -> [B,H,S,D] bf16, (v+b)*mask/scale
          v = (v + bias[cc]) * mask[rr] * scaleMul;
          int b = rr >> 12, s = rr & 4095, h = cc >> 6, d = cc & 63;
          outB[(((long)b * NH + h) * SEQ + s) * HD + d] = (bf16_t)v;
        } else if (EPI == 1) {   // V -> [B,H,D,S] bf16 (transposed)
          v = v + bias[cc];
          int b = rr >> 12, s = rr & 4095, h = cc >> 6, d = cc & 63;
          outB[(((long)b * NH + h) * HD + d) * SEQ + s] = (bf16_t)v;
        } else if (EPI == 2) {   // f32 + bias
          outF[(long)rr * N + cc] = v + bias[cc];
        } else if (EPI == 3) {   // k3 raw scores + additive mask, bf16
          v = v - 1e9f * (1.0f - mask[(long)(bz >> 4) * SEQ + cc]);
          outB[(long)bz * sOut + (long)rr * N + cc] = (bf16_t)v;
        } else if (EPI == 4) {   // acc * (1/rowsum)
          float sum = stats[((long)bz * 256 + rr) * 2 + 1];
          outF[(long)bz * sOut + (long)rr * N + cc] = v * (1.0f / sum);
        }
      }
    }
  }
}

// ---------------------------------------------------------------------------
// Batched f32 NN GEMM via bf16 WMMA: Out = eScale*(A*B') + eDiag*I,
// where B' = bScale*B + bDiag*I.  A[M x K] (lda=K), B[K x N] (ldb=N), f32.
// 128x256 block tile, 64x64 wave tiles. B staged transposed: one column per
// thread, 32 coalesced strided loads -> 4 packed ds_store_b128.
// ---------------------------------------------------------------------------
__global__ __launch_bounds__(256) void gemm_f32nn(
    const float* __restrict__ A, const float* __restrict__ B,
    float* __restrict__ Out,
    int N, int K, long sA, long sB, long sOut,
    float bScale, float bDiag, float eScale, float eDiag,
    int transStore, int ldOut)
{
  __shared__ bf16_t As[128][40];
  __shared__ bf16_t Bs[256][40];

  const int tid = threadIdx.x;
  const int l = tid & 31, w = tid >> 5;
  const int wm = w >> 2, wn = w & 3;
  const int g = l >> 4, rl = l & 15;
  const int bm = blockIdx.y, bn = blockIdx.x, bz = blockIdx.z;

  A += (long)bz * sA;
  B += (long)bz * sB;

  v8f acc[4][4];
#pragma unroll
  for (int mi = 0; mi < 4; ++mi)
#pragma unroll
    for (int ni = 0; ni < 4; ++ni) acc[mi][ni] = vzero8();

  const int r2 = tid >> 1;
  const int seg = (tid & 1) * 16;
  const int gnB = bn * 256 + tid;

  for (int k0 = 0; k0 < K; k0 += 32) {
    __syncthreads();
    {   // A stage: 16 consecutive f32 -> 16 bf16 -> two b128 LDS stores
      const float* ga = A + (long)(bm * 128 + r2) * K + k0 + seg;
      bf16_t tmp[16];
#pragma unroll
      for (int i = 0; i < 16; ++i) tmp[i] = (bf16_t)ga[i];
      *(uint4*)&As[r2][seg]     = *(const uint4*)&tmp[0];
      *(uint4*)&As[r2][seg + 8] = *(const uint4*)&tmp[8];
    }
    {   // B stage: one column per thread, transform + pack + 4 b128 stores
      bf16_t tmp[32];
      if (gnB < N) {
        const float* gb = B + (long)k0 * N + gnB;
#pragma unroll
        for (int i = 0; i < 32; ++i) {
          float x = gb[(long)i * N];
          x = bScale * x + (((k0 + i) == gnB) ? bDiag : 0.0f);
          tmp[i] = (bf16_t)x;
        }
      } else {
#pragma unroll
        for (int i = 0; i < 32; ++i) tmp[i] = (bf16_t)0.0f;
      }
#pragma unroll
      for (int c = 0; c < 4; ++c)
        *(uint4*)&Bs[tid][c * 8] = *(const uint4*)&tmp[c * 8];
    }
    __syncthreads();

    v16bf afr[4], bfr[4];
#pragma unroll
    for (int mi = 0; mi < 4; ++mi) {
      FragB16 f;
      f.q[0] = *(const uint4*)&As[wm * 64 + mi * 16 + rl][g * 8];
      f.q[1] = *(const uint4*)&As[wm * 64 + mi * 16 + rl][16 + g * 8];
      afr[mi] = f.v;
    }
#pragma unroll
    for (int ni = 0; ni < 4; ++ni) {
      FragB16 f;
      f.q[0] = *(const uint4*)&Bs[wn * 64 + ni * 16 + rl][g * 8];
      f.q[1] = *(const uint4*)&Bs[wn * 64 + ni * 16 + rl][16 + g * 8];
      bfr[ni] = f.v;
    }
#pragma unroll
    for (int mi = 0; mi < 4; ++mi)
#pragma unroll
      for (int ni = 0; ni < 4; ++ni)
        acc[mi][ni] = wmma_bf(afr[mi], bfr[ni], acc[mi][ni]);
  }

#pragma unroll
  for (int mi = 0; mi < 4; ++mi) {
#pragma unroll
    for (int ni = 0; ni < 4; ++ni) {
#pragma unroll
      for (int e = 0; e < 8; ++e) {
        int rr = bm * 128 + wm * 64 + mi * 16 + g * 8 + e;
        int cc = bn * 256 + wn * 64 + ni * 16 + rl;
        if (cc >= N) continue;
        float v = eScale * acc[mi][ni][e] + ((rr == cc) ? eDiag : 0.0f);
        long off = (long)bz * sOut +
                   (transStore ? ((long)cc * ldOut + rr)
                               : ((long)rr * ldOut + cc));
        Out[off] = v;
      }
    }
  }
}

// ---------------------------------------------------------------------------
// k2 = softmax(Ql @ Kl^T): one block per (band of 16 landmark rows, bh).
// ---------------------------------------------------------------------------
__global__ __launch_bounds__(256) void k2_softmax(
    const bf16_t* __restrict__ Qlm, const bf16_t* __restrict__ Klm,
    float* __restrict__ k2)
{
  __shared__ bf16_t Aq[16][72];
  __shared__ float  sc[16][264];
  const int tid = threadIdx.x, l = tid & 31, w = tid >> 5;
  const int band = blockIdx.x, bh = blockIdx.y;
  const long lbase = (long)bh * NLM * HD;
  const int g = l >> 4, rl = l & 15;

#pragma unroll
  for (int i = 0; i < 4; ++i) {
    int e = tid * 4 + i; int r = e >> 6, c = e & 63;
    Aq[r][c] = Qlm[lbase + (long)(band * 16 + r) * HD + c];
  }
  __syncthreads();

#pragma unroll
  for (int ni = 0; ni < 2; ++ni) {
    int col0 = w * 32 + ni * 16;
    v8f acc = vzero8();
#pragma unroll
    for (int ks = 0; ks < 2; ++ks) {
      int k0 = ks * 32;
      FragB16 fa, fb;
      fa.q[0] = *(const uint4*)&Aq[rl][k0 + g * 8];
      fa.q[1] = *(const uint4*)&Aq[rl][k0 + 16 + g * 8];
      const bf16_t* kb = Klm + lbase + (long)(col0 + rl) * HD + k0;
      fb.q[0] = *(const uint4*)(kb + g * 8);
      fb.q[1] = *(const uint4*)(kb + 16 + g * 8);
      acc = wmma_bf(fa.v, fb.v, acc);
    }
#pragma unroll
    for (int e = 0; e < 8; ++e) sc[g * 8 + e][col0 + rl] = acc[e];
  }
  __syncthreads();

#pragma unroll
  for (int rr = 0; rr < 2; ++rr) {
    int r = w * 2 + rr;
    float mx = -3.0e38f;
    for (int j = l; j < NLM; j += 32) mx = fmaxf(mx, sc[r][j]);
#pragma unroll
    for (int o = 16; o; o >>= 1) mx = fmaxf(mx, __shfl_xor(mx, o, 32));
    float s = 0.0f;
    for (int j = l; j < NLM; j += 32) {
      float e = __expf(sc[r][j] - mx); sc[r][j] = e; s += e;
    }
#pragma unroll
    for (int o = 16; o; o >>= 1) s += __shfl_xor(s, o, 32);
    float inv = 1.0f / s;
    for (int j = l; j < NLM; j += 32)
      k2[(long)bh * NLM * NLM + (long)(band * 16 + r) * NLM + j] = sc[r][j] * inv;
  }
}

// ---------------------------------------------------------------------------
// Fused k1 + final attention band:
//   scores = Q_band @ Klm^T ; softmax ; out = scores @ M2  (M2t = (inv@k3V)^T)
// One block per (16-row seq band, bh); writes attn [B,S,H*D] bf16.
// ---------------------------------------------------------------------------
__global__ __launch_bounds__(256) void k1_fused(
    const bf16_t* __restrict__ Qbf, const bf16_t* __restrict__ Klm,
    const float* __restrict__ M2t, bf16_t* __restrict__ attn)
{
  __shared__ bf16_t Aq[16][72];
  __shared__ float  sc[16][264];
  __shared__ float  part[2][16][64];
  __shared__ float  rinv[16];
  const int tid = threadIdx.x, l = tid & 31, w = tid >> 5;
  const int band = blockIdx.x, bh = blockIdx.y;
  const int s0 = band * 16;
  const long qbase = (long)bh * SEQ * HD;
  const long lbase = (long)bh * NLM * HD;
  const int g = l >> 4, rl = l & 15;

#pragma unroll
  for (int i = 0; i < 4; ++i) {
    int e = tid * 4 + i; int r = e >> 6, c = e & 63;
    Aq[r][c] = Qbf[qbase + (long)(s0 + r) * HD + c];
  }
  __syncthreads();

#pragma unroll
  for (int ni = 0; ni < 2; ++ni) {
    int col0 = w * 32 + ni * 16;
    v8f acc = vzero8();
#pragma unroll
    for (int ks = 0; ks < 2; ++ks) {
      int k0 = ks * 32;
      FragB16 fa, fb;
      fa.q[0] = *(const uint4*)&Aq[rl][k0 + g * 8];
      fa.q[1] = *(const uint4*)&Aq[rl][k0 + 16 + g * 8];
      const bf16_t* kb = Klm + lbase + (long)(col0 + rl) * HD + k0;
      fb.q[0] = *(const uint4*)(kb + g * 8);
      fb.q[1] = *(const uint4*)(kb + 16 + g * 8);
      acc = wmma_bf(fa.v, fb.v, acc);
    }
#pragma unroll
    for (int e = 0; e < 8; ++e) sc[g * 8 + e][col0 + rl] = acc[e];
  }
  __syncthreads();

#pragma unroll
  for (int rr = 0; rr < 2; ++rr) {
    int r = w * 2 + rr;
    float mx = -3.0e38f;
    for (int j = l; j < NLM; j += 32) mx = fmaxf(mx, sc[r][j]);
#pragma unroll
    for (int o = 16; o; o >>= 1) mx = fmaxf(mx, __shfl_xor(mx, o, 32));
    float s = 0.0f;
    for (int j = l; j < NLM; j += 32) {
      float e = __expf(sc[r][j] - mx); sc[r][j] = e; s += e;
    }
#pragma unroll
    for (int o = 16; o; o >>= 1) s += __shfl_xor(s, o, 32);
    if (l == 0) rinv[r] = 1.0f / s;
  }
  __syncthreads();

  { // [16 x 256] @ [256 x 64]; waves: 4 col tiles x 2 K halves
    int ni = w & 3, kh = w >> 2;
    v8f acc = vzero8();
    const float* bp = M2t + (long)bh * HD * NLM + (long)(ni * 16 + rl) * NLM;
#pragma unroll
    for (int ks = 0; ks < 4; ++ks) {
      int k0 = kh * 128 + ks * 32;
      v16bf a, b;
#pragma unroll
      for (int i = 0; i < 8; ++i) {
        a[i]     = (bf16_t)sc[rl][k0 + g * 8 + i];
        a[8 + i] = (bf16_t)sc[rl][k0 + 16 + g * 8 + i];
      }
#pragma unroll
      for (int i = 0; i < 8; ++i) {
        b[i]     = (bf16_t)bp[k0 + g * 8 + i];
        b[8 + i] = (bf16_t)bp[k0 + 16 + g * 8 + i];
      }
      acc = wmma_bf(a, b, acc);
    }
#pragma unroll
    for (int e = 0; e < 8; ++e) part[kh][g * 8 + e][ni * 16 + rl] = acc[e];
  }
  __syncthreads();

  {
    int b = bh >> 4, h = bh & 15;
#pragma unroll
    for (int e = 0; e < 4; ++e) {
      int i = tid + e * 256;
      int r = i >> 6, c = i & 63;
      float v = (part[0][r][c] + part[1][r][c]) * rinv[r];
      attn[((long)b * SEQ + s0 + r) * (NH * HD) + h * HD + c] = (bf16_t)v;
    }
  }
}

// ---------------------------------------------------------------------------
// Small helper kernels
// ---------------------------------------------------------------------------
__global__ void cvt_bf16(const float* __restrict__ s, bf16_t* __restrict__ d)
{
  long i = (long)blockIdx.x * 256 + threadIdx.x;
  d[i] = (bf16_t)s[i];
}

__global__ void transpose_cvt(const float* __restrict__ src,
                              bf16_t* __restrict__ dst) // 1024x1024 -> T bf16
{
  __shared__ float t[32][33];
  int x = blockIdx.x * 32 + threadIdx.x;
#pragma unroll
  for (int j = 0; j < 4; ++j) {
    int y = blockIdx.y * 32 + threadIdx.y + j * 8;
    t[threadIdx.y + j * 8][threadIdx.x] = src[(long)y * 1024 + x];
  }
  __syncthreads();
  int x2 = blockIdx.y * 32 + threadIdx.x;
#pragma unroll
  for (int j = 0; j < 4; ++j) {
    int y2 = blockIdx.x * 32 + threadIdx.y + j * 8;
    dst[(long)y2 * 1024 + x2] = (bf16_t)t[threadIdx.x][threadIdx.y + j * 8];
  }
}

__global__ void landmark_mean(const bf16_t* __restrict__ src,
                              bf16_t* __restrict__ dst)
{
  long idx = (long)blockIdx.x * 256 + threadIdx.x;   // over BH*NLM*HD
  int d = (int)(idx & 63);
  long rem = idx >> 6;
  int lm = (int)(rem & 255);
  long bh = rem >> 8;
  const bf16_t* p = src + bh * (long)SEQ * HD + (long)lm * 16 * HD + d;
  float s = 0.0f;
#pragma unroll
  for (int t = 0; t < 16; ++t) s += (float)p[t * HD];
  dst[idx] = (bf16_t)(s * (1.0f / 16.0f));
}

__global__ void init_denom(float* __restrict__ denom)
{
  if (threadIdx.x < 2) denom[threadIdx.x] = 0.0f;
}

__global__ __launch_bounds__(256) void k2_absmax(const float* __restrict__ k2,
                                                 float* __restrict__ denom)
{
  __shared__ float redc[8], redr[8];
  const int tid = threadIdx.x, l = tid & 31, w = tid >> 5;
  const float* p = k2 + (long)blockIdx.x * NLM * NLM;
  float cs = 0.0f, rs = 0.0f;
  for (int i = 0; i < NLM; ++i) cs += fabsf(p[(long)i * NLM + tid]);
  for (int j = 0; j < NLM; ++j) rs += fabsf(p[(long)tid * NLM + j]);
#pragma unroll
  for (int o = 16; o; o >>= 1) {
    cs = fmaxf(cs, __shfl_xor(cs, o, 32));
    rs = fmaxf(rs, __shfl_xor(rs, o, 32));
  }
  if (l == 0) { redc[w] = cs; redr[w] = rs; }
  __syncthreads();
  if (tid == 0) {
    float c = redc[0], r = redr[0];
    for (int i = 1; i < 8; ++i) { c = fmaxf(c, redc[i]); r = fmaxf(r, redr[i]); }
    atomicMax((int*)&denom[0], __float_as_int(c));
    atomicMax((int*)&denom[1], __float_as_int(r));
  }
}

__global__ void newton_init(const float* __restrict__ k2,
                            const float* __restrict__ denom,
                            float* __restrict__ V)
{
  long idx = (long)blockIdx.x * 256 + threadIdx.x;  // over BH*256*256
  long bh = idx >> 16;
  int rem = (int)(idx & 65535);
  int i = rem >> 8, j = rem & 255;
  float rd = 1.0f / (denom[0] * denom[1]);
  V[idx] = k2[(bh << 16) + ((long)j << 8) + i] * rd;
}

__global__ __launch_bounds__(256) void rowstats(const bf16_t* __restrict__ k3raw,
                                                float* __restrict__ stats)
{
  const int l = threadIdx.x & 31, w = threadIdx.x >> 5;
  long row = (long)blockIdx.x * 8 + w;       // over BH*NLM = 16384 rows
  const bf16_t* p = k3raw + row * SEQ;
  float mx = -3.0e38f;
  for (int j = l; j < SEQ; j += 32) mx = fmaxf(mx, (float)p[j]);
#pragma unroll
  for (int o = 16; o; o >>= 1) mx = fmaxf(mx, __shfl_xor(mx, o, 32));
  float s = 0.0f;
  for (int j = l; j < SEQ; j += 32) s += __expf((float)p[j] - mx);
#pragma unroll
  for (int o = 16; o; o >>= 1) s += __shfl_xor(s, o, 32);
  if (l == 0) { stats[row * 2] = mx; stats[row * 2 + 1] = s; }
}

// ---------------------------------------------------------------------------
// Host-side orchestration
// ---------------------------------------------------------------------------
extern "C" void kernel_launch(void* const* d_in, const int* in_sizes, int n_in,
                              void* d_out, int out_size, void* d_ws, size_t ws_size,
                              hipStream_t stream)
{
  const float* X    = (const float*)d_in[0];
  const float* mask = (const float*)d_in[1];
  const float* Wq   = (const float*)d_in[2];
  const float* bq   = (const float*)d_in[3];
  const float* Wk   = (const float*)d_in[4];
  const float* bk   = (const float*)d_in[5];
  const float* Wv   = (const float*)d_in[6];
  const float* bv   = (const float*)d_in[7];
  const float* Wff  = (const float*)d_in[8];
  const float* bff  = (const float*)d_in[9];
  float* out = (float*)d_out;

  char* base = (char*)d_ws;
  size_t cur = 0;
  auto alloc = [&](size_t bytes) -> char* {
    char* p = base + cur;
    cur = (cur + bytes + 255) & ~(size_t)255;
    return p;
  };

  const long MROWS = (long)NBAT * SEQ;               // 16384
  bf16_t* Xbf   = (bf16_t*)alloc((size_t)MROWS * 1024 * 2);
  bf16_t* WqT   = (bf16_t*)alloc((size_t)1024 * 1024 * 2);
  bf16_t* WkT   = (bf16_t*)alloc((size_t)1024 * 1024 * 2);
  bf16_t* WvT   = (bf16_t*)alloc((size_t)1024 * 1024 * 2);
  bf16_t* WffT  = (bf16_t*)alloc((size_t)1024 * 1024 * 2);
  bf16_t* Qbf   = (bf16_t*)alloc((size_t)BH * SEQ * HD * 2);
  bf16_t* Kbf   = (bf16_t*)alloc((size_t)BH * SEQ * HD * 2);
  bf16_t* Vt    = (bf16_t*)alloc((size_t)BH * HD * SEQ * 2);
  bf16_t* Qlm   = (bf16_t*)alloc((size_t)BH * NLM * HD * 2);
  bf16_t* Klm   = (bf16_t*)alloc((size_t)BH * NLM * HD * 2);
  float*  k2f   = (float*) alloc((size_t)BH * NLM * NLM * 4);
  float*  KV    = (float*) alloc((size_t)BH * NLM * NLM * 4);
  float*  Ubuf  = (float*) alloc((size_t)BH * NLM * NLM * 4);
  float*  U2buf = (float*) alloc((size_t)BH * NLM * NLM * 4);
  float*  Vping = (float*) alloc((size_t)BH * NLM * NLM * 4);
  float*  Vpong = (float*) alloc((size_t)BH * NLM * NLM * 4);
  bf16_t* k3raw = (bf16_t*)alloc((size_t)BH * NLM * SEQ * 2);
  float*  stats = (float*) alloc((size_t)BH * NLM * 2 * 4);
  float*  k3V   = (float*) alloc((size_t)BH * NLM * HD * 4);
  float*  M2t   = (float*) alloc((size_t)BH * HD * NLM * 4);
  bf16_t* attn  = (bf16_t*)alloc((size_t)MROWS * 1024 * 2);
  float*  denom = (float*) alloc(256);
  (void)ws_size; (void)in_sizes; (void)n_in; (void)out_size;

  const float invScale = 0.35355339059327373f;   // 1 / 64^0.25

  // --- precision conversion / weight transposition ---
  cvt_bf16<<<65536, 256, 0, stream>>>(X, Xbf);
  transpose_cvt<<<dim3(32, 32), dim3(32, 8), 0, stream>>>(Wq,  WqT);
  transpose_cvt<<<dim3(32, 32), dim3(32, 8), 0, stream>>>(Wk,  WkT);
  transpose_cvt<<<dim3(32, 32), dim3(32, 8), 0, stream>>>(Wv,  WvT);
  transpose_cvt<<<dim3(32, 32), dim3(32, 8), 0, stream>>>(Wff, WffT);
  init_denom<<<1, 32, 0, stream>>>(denom);

  // --- QKV projections (WMMA, async-staged, 128x256 blocks) ---
  gemm_bf16_nt<0, false><<<dim3(4, 128, 1), 256, 0, stream>>>(
      Xbf, WqT, 1024, 1024, 1024, 1024, 0, 0, 0,
      bq, mask, nullptr, nullptr, Qbf, invScale);
  gemm_bf16_nt<0, false><<<dim3(4, 128, 1), 256, 0, stream>>>(
      Xbf, WkT, 1024, 1024, 1024, 1024, 0, 0, 0,
      bk, mask, nullptr, nullptr, Kbf, invScale);
  gemm_bf16_nt<1, false><<<dim3(4, 128, 1), 256, 0, stream>>>(
      Xbf, WvT, 1024, 1024, 1024, 1024, 0, 0, 0,
      bv, mask, nullptr, nullptr, Vt, 1.0f);

  // --- landmarks ---
  landmark_mean<<<4096, 256, 0, stream>>>(Qbf, Qlm);
  landmark_mean<<<4096, 256, 0, stream>>>(Kbf, Klm);

  // --- k2 softmax + Newton-Schulz setup ---
  k2_softmax<<<dim3(16, 64), 256, 0, stream>>>(Qlm, Klm, k2f);
  k2_absmax<<<64, 256, 0, stream>>>(k2f, denom);
  newton_init<<<16384, 256, 0, stream>>>(k2f, denom, Vping);

  // --- k3 raw scores (bf16), row stats, k3 @ V with fused softmax ---
  gemm_bf16_nt<3, false><<<dim3(16, 2, 64), 256, 0, stream>>>(
      Qlm, Kbf, SEQ, HD, HD, HD,
      (long)NLM * HD, (long)SEQ * HD, (long)NLM * SEQ,
      nullptr, mask, nullptr, nullptr, k3raw, 1.0f);
  rowstats<<<2048, 256, 0, stream>>>(k3raw, stats);
  gemm_bf16_nt<4, true><<<dim3(1, 2, 64), 256, 0, stream>>>(
      k3raw, Vt, HD, SEQ, SEQ, SEQ,
      (long)NLM * SEQ, (long)HD * SEQ, (long)NLM * HD,
      nullptr, nullptr, stats, k3V, nullptr, 1.0f);

  // --- Newton-Schulz pseudo-inverse: 6 iterations, 4 GEMMs each ---
  float* Vp = Vping;
  float* Vq = Vpong;
  for (int it = 0; it < 6; ++it) {
    gemm_f32nn<<<dim3(1, 2, 64), 256, 0, stream>>>(
        k2f, Vp, KV, 256, 256, 65536, 65536, 65536,
        1.0f, 0.0f, 1.0f, 0.0f, 0, 256);                  // KV = K2 @ V
    gemm_f32nn<<<dim3(1, 2, 64), 256, 0, stream>>>(
        KV, KV, Ubuf, 256, 256, 65536, 65536, 65536,
        -1.0f, 7.0f, -1.0f, 15.0f, 0, 256);               // U = 15I - KV(7I-KV)
    gemm_f32nn<<<dim3(1, 2, 64), 256, 0, stream>>>(
        KV, Ubuf, U2buf, 256, 256, 65536, 65536, 65536,
        1.0f, 0.0f, -1.0f, 13.0f, 0, 256);                // U2 = 13I - KV@U
    gemm_f32nn<<<dim3(1, 2, 64), 256, 0, stream>>>(
        Vp, U2buf, Vq, 256, 256, 65536, 65536, 65536,
        1.0f, 0.0f, 0.25f, 0.0f, 0, 256);                 // V = 0.25 V @ U2
    float* t = Vp; Vp = Vq; Vq = t;
  }

  // --- M2t = (inv @ k3V)^T  [BH, 64, 256] ---
  gemm_f32nn<<<dim3(1, 2, 64), 256, 0, stream>>>(
      Vp, k3V, M2t, 64, 256, 65536, (long)NLM * HD, (long)HD * NLM,
      1.0f, 0.0f, 1.0f, 0.0f, 1, 256);

  // --- fused k1 softmax + (k1 @ M2) -> attn [B,S,H*D] bf16 ---
  k1_fused<<<dim3(256, 64), 256, 0, stream>>>(Qbf, Klm, M2t, attn);

  // --- final projection: out = attn @ Wff + bff (f32) ---
  gemm_bf16_nt<2, false><<<dim3(4, 128, 1), 256, 0, stream>>>(
      attn, WffT, 1024, 1024, 1024, 1024, 0, 0, 0,
      bff, nullptr, nullptr, out, nullptr, 1.0f);
}
